// CSPNet_50302656971429
// MI455X (gfx1250) — compile-verified, matrix operand order checked by reading, not run on hardware
//
#include <hip/hip_runtime.h>
#include <hip/hip_bf16.h>

typedef __attribute__((ext_vector_type(16))) _Float16 v16h;
typedef __attribute__((ext_vector_type(8)))  _Float16 v8h;
typedef __attribute__((ext_vector_type(8)))  float    v8f;
typedef __attribute__((ext_vector_type(4)))  int      v4i;

#define GG 512
#define AA 24
#define NN (GG*AA)       // 12288 nodes
#define HH 128
#define EPG (AA*AA)      // 576 edges per graph
#define EE (GG*EPG)      // 294912 edges
#define NFQ 10
#define K1PAD 352        // 325 -> 11 K-tiles of 32
#define CH 64            // edge chunk (4 M-tiles)
#define NCHUNK (EPG/CH)  // 9

// ---- CDNA5 async global->LDS copy (guarded; falls back to sync copy) ------
#if __has_builtin(__builtin_amdgcn_global_load_async_to_lds_b128) && \
    __has_builtin(__builtin_amdgcn_s_wait_asynccnt)
#define HAVE_ASYNC_LDS 1
#else
#define HAVE_ASYNC_LDS 0
#endif

typedef __attribute__((address_space(1))) v4i glb_v4i;   // printed by clang as "__device__"
typedef __attribute__((address_space(3))) v4i lds_v4i;   // printed by clang as "__shared__"

__device__ __forceinline__ void async_copy16(const void* gsrc, void* ldst) {
#if HAVE_ASYNC_LDS
  // generic->AS1: identical 64-bit representation; generic->AS3: low 32 bits
  // are the LDS offset (ISA 10.2). Integer casts avoid
  // illegal addrspace pointer casts.
  __builtin_amdgcn_global_load_async_to_lds_b128(
      (glb_v4i*)(uintptr_t)gsrc,
      (lds_v4i*)(uint32_t)(uintptr_t)ldst, 0, 0);
#else
  *(v8h*)ldst = *(const v8h*)gsrc;
#endif
}

__device__ __forceinline__ void async_fence() {
#if HAVE_ASYNC_LDS
  __builtin_amdgcn_s_wait_asynccnt(0);
#endif
}

__device__ __forceinline__ float silu_f(float x) { return x / (1.0f + __expf(-x)); }

// Build a 16-half WMMA fragment from two contiguous 8-half (16B) pieces:
// per the CDNA5 16-bit layout a lane holds K = kb..kb+7 and kb+16..kb+23.
__device__ __forceinline__ v16h make_frag(const _Float16* lo) {
  v8h a = *(const v8h*)(lo);
  v8h b = *(const v8h*)(lo + 16);
  v16h r;
  #pragma unroll
  for (int i = 0; i < 8; ++i) { r[i] = a[i]; r[8 + i] = b[i]; }
  return r;
}

// ---------------------------------------------------------------------------
// Fused per-layer kernel: one workgroup (8 wave32) per graph.
//   edge MLP (WMMA f16) -> scatter-mean (ds_add_f32) -> node MLP (WMMA) -> h+=
// e_in K layout (permuted, weights permuted to match):
//   [0..127]=h_src  [128..255]=h_dst  [256..319]=fdemb(60+4 pad)
//   [320..328]=lat_ips(9)  [329..351]=0
// ---------------------------------------------------------------------------
__global__ __launch_bounds__(256, 1)
void layer_kernel(float* __restrict__ h,                 // [N,128] in/out (in-place safe per graph)
                  const _Float16* __restrict__ fdemb,    // [E,64] f16 (60 valid)
                  const float* __restrict__ latips,      // [G,9]
                  const _Float16* __restrict__ W1t,      // [128,352] (out-major, K permuted+padded)
                  const float* __restrict__ B1,          // [128]
                  const _Float16* __restrict__ W2t,      // [128,128]
                  const float* __restrict__ B2,
                  const _Float16* __restrict__ N1t,      // [128,256]
                  const float* __restrict__ NB1,
                  const _Float16* __restrict__ N2t,      // [128,128]
                  const float* __restrict__ NB2)
{
  __shared__ __align__(16) _Float16 hloc[32][HH];   // 8 KB  node features (rows 24..31 = 0)
  __shared__ __align__(16) _Float16 tail[CH][96];   // 12 KB fdemb64 | lat9 | pad (K-tiles 8..10)
  __shared__ __align__(16) _Float16 ef1[CH][HH];    // 16 KB hidden acts (reused by node MLP)
  __shared__ __align__(16) float    agg[32][HH];    // 16 KB scatter-sum accumulator

  const int g    = blockIdx.x;
  const int tid  = threadIdx.x;
  const int lane = tid & 31;
  const int hl   = lane >> 4;       // half-wave select (A/B/C layout)
  const int l16  = lane & 15;
  const int kb   = hl * 8;          // K sub-offset (halves)
  const int nt   = tid >> 5;        // wave -> N-tile (8 waves, 8 N-tiles)
  const int ncol = nt * 16 + l16;   // output column owned by this lane

  __builtin_prefetch(W1t, 0, 3);
  __builtin_prefetch(W2t, 0, 3);
  __builtin_prefetch(N1t, 0, 3);
  __builtin_prefetch(N2t, 0, 3);

  // Stage node features (f32 -> f16), zero aggregator.
  for (int idx = tid; idx < 32 * HH; idx += 256) {
    int r = idx >> 7, c = idx & 127;
    hloc[r][c] = (r < AA) ? (_Float16)h[(size_t)(g * AA + r) * HH + c] : (_Float16)0.0f;
    agg[r][c] = 0.0f;
  }
  // Chunk-invariant tail columns: [64..72]=lat_ips, [73..95]=0.
  for (int idx = tid; idx < CH * 32; idx += 256) {
    int r = idx >> 5, c = idx & 31;
    tail[r][64 + c] = (c < 9) ? (_Float16)latips[g * 9 + c] : (_Float16)0.0f;
  }
  __syncthreads();

  // -------------------- edge phase: 9 chunks of 64 edges --------------------
  for (int chunk = 0; chunk < NCHUNK; ++chunk) {
    const int e0 = chunk * CH;

    // Async-stage fdemb block: tail[r][0..63] <= fdemb[e0+r] (128 B per row,
    // 512 x 16B segments, 2 per thread) via global_load_async_to_lds_b128.
    for (int s = tid; s < CH * 8; s += 256) {
      int r = s >> 3, c = (s & 7) * 8;
      async_copy16(fdemb + (size_t)(g * EPG + e0 + r) * 64 + c, &tail[r][c]);
    }
    async_fence();                      // s_wait_asynccnt 0
    __syncthreads();

    // GEMM1: ef1 = silu(e_in @ W1 + B1)   M=64, K=352, N=128
    #pragma unroll
    for (int mt = 0; mt < CH / 16; ++mt) {
      const int arow  = mt * 16 + l16;       // A-matrix row (edge in chunk)
      const int le    = e0 + arow;
      const int inode = le / AA;             // src node (e_in cols 0..127)
      const int jnode = le % AA;             // dst node (e_in cols 128..255)
      v8f acc = {};
      #pragma unroll
      for (int kt = 0; kt < 11; ++kt) {
        const _Float16* ap;
        if (kt < 4)      ap = &hloc[inode][kt * 32 + kb];
        else if (kt < 8) ap = &hloc[jnode][(kt - 4) * 32 + kb];
        else             ap = &tail[arow][(kt - 8) * 32 + kb];
        v16h afrag = make_frag(ap);
        v16h bfrag = make_frag(W1t + (size_t)ncol * K1PAD + kt * 32 + kb);
        acc = __builtin_amdgcn_wmma_f32_16x16x32_f16(false, afrag, false, bfrag,
                                                     (short)0, acc, false, false);
      }
      const float bias = B1[ncol];
      #pragma unroll
      for (int r = 0; r < 8; ++r) {
        int mrow = mt * 16 + r + hl * 8;     // C-layout row
        ef1[mrow][ncol] = (_Float16)silu_f(acc[r] + bias);
      }
    }
    __syncthreads();

    // GEMM2 + scatter-mean: ef2 = silu(ef1 @ W2 + B2); agg[src] += ef2
    #pragma unroll
    for (int mt = 0; mt < CH / 16; ++mt) {
      const int arow = mt * 16 + l16;
      v8f acc = {};
      #pragma unroll
      for (int kt = 0; kt < 4; ++kt) {
        v16h afrag = make_frag(&ef1[arow][kt * 32 + kb]);
        v16h bfrag = make_frag(W2t + (size_t)ncol * HH + kt * 32 + kb);
        acc = __builtin_amdgcn_wmma_f32_16x16x32_f16(false, afrag, false, bfrag,
                                                     (short)0, acc, false, false);
      }
      const float bias = B2[ncol];
      #pragma unroll
      for (int r = 0; r < 8; ++r) {
        int mrow = mt * 16 + r + hl * 8;
        int le   = e0 + mrow;
        atomicAdd(&agg[le / AA][ncol], silu_f(acc[r] + bias));   // ds_add_f32
      }
    }
    __syncthreads();
  }

  // -------------------- node phase: M=32 (24 valid), K=256 ------------------
  const float invdeg = 1.0f / (float)AA;
  #pragma unroll
  for (int mt = 0; mt < 2; ++mt) {
    const int arow = mt * 16 + l16;
    v8f acc = {};
    #pragma unroll
    for (int kt = 0; kt < 8; ++kt) {
      v16h afrag;
      if (kt < 4) {
        afrag = make_frag(&hloc[arow][kt * 32 + kb]);
      } else {                               // x cols 128..255 = agg/deg (f32 -> f16)
        const float* fp = &agg[arow][(kt - 4) * 32 + kb];
        #pragma unroll
        for (int x = 0; x < 8; ++x) {
          afrag[x]     = (_Float16)(fp[x]      * invdeg);
          afrag[8 + x] = (_Float16)(fp[16 + x] * invdeg);
        }
      }
      v16h bfrag = make_frag(N1t + (size_t)ncol * 256 + kt * 32 + kb);
      acc = __builtin_amdgcn_wmma_f32_16x16x32_f16(false, afrag, false, bfrag,
                                                   (short)0, acc, false, false);
    }
    const float bias = NB1[ncol];
    #pragma unroll
    for (int r = 0; r < 8; ++r) {
      int mrow = mt * 16 + r + hl * 8;
      ef1[mrow][ncol] = (_Float16)silu_f(acc[r] + bias);
    }
  }
  __syncthreads();
  #pragma unroll
  for (int mt = 0; mt < 2; ++mt) {
    const int arow = mt * 16 + l16;
    v8f acc = {};
    #pragma unroll
    for (int kt = 0; kt < 4; ++kt) {
      v16h afrag = make_frag(&ef1[arow][kt * 32 + kb]);
      v16h bfrag = make_frag(N2t + (size_t)ncol * HH + kt * 32 + kb);
      acc = __builtin_amdgcn_wmma_f32_16x16x32_f16(false, afrag, false, bfrag,
                                                   (short)0, acc, false, false);
    }
    const float bias = NB2[ncol];
    #pragma unroll
    for (int r = 0; r < 8; ++r) {
      int mrow = mt * 16 + r + hl * 8;
      if (mrow < AA) {                       // residual: h += out
        size_t o = (size_t)(g * AA + mrow) * HH + ncol;
        h[o] = h[o] + silu_f(acc[r] + bias);
      }
    }
  }
}

// ----------------------------- prep kernels --------------------------------
__global__ void latent_kernel(const float* __restrict__ t, const int* __restrict__ at,
                              const float* __restrict__ xrd, const float* __restrict__ emb,
                              const float* __restrict__ W, const float* __restrict__ b,
                              float* __restrict__ h)
{
  int n = blockIdx.x, c = threadIdx.x;
  int g = n / AA;
  int a = at[n] - 1;
  float acc = b[c];
  const float* er = emb + (size_t)a * HH;
  for (int k = 0; k < HH; ++k)  acc += er[k] * W[(size_t)k * HH + c];
  const float* tr = t + (size_t)g * 128;
  for (int k = 0; k < 128; ++k) acc += tr[k] * W[(size_t)(HH + k) * HH + c];
  const float* xr = xrd + (size_t)g * 128;
  for (int k = 0; k < 128; ++k) acc += xr[k] * W[(size_t)(256 + k) * HH + c];
  h[(size_t)n * HH + c] = acc;
}

__global__ void fdemb_kernel(const float* __restrict__ frac, _Float16* __restrict__ fdemb)
{
  int e = blockIdx.x * blockDim.x + threadIdx.x;
  if (e >= EE) return;
  int g = e / EPG, le = e % EPG;
  int i = le / AA, j = le % AA;
  int src = g * AA + i, dst = g * AA + j;
  _Float16* out = fdemb + (size_t)e * 64;
  const float TWOPI = 6.283185307179586f;
  #pragma unroll
  for (int d = 0; d < 3; ++d) {
    float diff = frac[dst * 3 + d] - frac[src * 3 + d];
    diff = diff - floorf(diff);                          // mod 1.0
    #pragma unroll
    for (int f = 0; f < NFQ; ++f) {
      float ang = diff * (TWOPI * (float)f);
      out[d * NFQ + f]      = (_Float16)__sinf(ang);
      out[30 + d * NFQ + f] = (_Float16)__cosf(ang);
    }
  }
  out[60] = out[61] = out[62] = out[63] = (_Float16)0.0f;
}

__global__ void latips_kernel(const float* __restrict__ lat, float* __restrict__ ips)
{
  int idx = blockIdx.x * blockDim.x + threadIdx.x;
  if (idx >= GG * 9) return;
  int g = idx / 9, r = idx % 9;
  int i = r / 3, j = r % 3;
  const float* L = lat + (size_t)g * 9;
  ips[idx] = L[i*3+0]*L[j*3+0] + L[i*3+1]*L[j*3+1] + L[i*3+2]*L[j*3+2];
}

// edge_w1 [L,325,128] -> [L,128,352] f16 with the e_in K permutation:
//   dst k: [0..255]=src k     [256..315]=src 265+(k-256) (fdemb)
//          [316..319]=0       [320..328]=src 256+(k-320) (lat9)   [329..351]=0
__global__ void conv_w1_kernel(const float* __restrict__ src, _Float16* __restrict__ dst)
{
  int idx = blockIdx.x * blockDim.x + threadIdx.x;
  if (idx >= 4 * HH * K1PAD) return;
  int l = idx / (HH * K1PAD);
  int rem = idx % (HH * K1PAD);
  int n = rem / K1PAD, k = rem % K1PAD;
  int s;
  if (k < 256)      s = k;
  else if (k < 316) s = 265 + (k - 256);
  else if (k < 320) s = -1;
  else if (k < 329) s = 256 + (k - 320);
  else              s = -1;
  dst[idx] = (s >= 0) ? (_Float16)src[((size_t)l * 325 + s) * HH + n] : (_Float16)0.0f;
}

// generic [L,K,128] -> [L,128,K] f16
__global__ void conv_wt_kernel(const float* __restrict__ src, _Float16* __restrict__ dst, int K)
{
  int idx = blockIdx.x * blockDim.x + threadIdx.x;
  if (idx >= 4 * HH * K) return;
  int l = idx / (HH * K);
  int rem = idx % (HH * K);
  int n = rem / K, k = rem % K;
  dst[idx] = (_Float16)src[((size_t)l * K + k) * HH + n];
}

// ----------------------------- output kernels ------------------------------
__global__ void coord_kernel(const float* __restrict__ h, const float* __restrict__ cw,
                             float* __restrict__ out)
{
  int n = blockIdx.x * blockDim.x + threadIdx.x;
  if (n >= NN) return;
  float a0 = 0.f, a1 = 0.f, a2 = 0.f;
  const float* hr = h + (size_t)n * HH;
  for (int k = 0; k < HH; ++k) {
    float v = hr[k];
    a0 += v * cw[k * 3 + 0]; a1 += v * cw[k * 3 + 1]; a2 += v * cw[k * 3 + 2];
  }
  out[n * 3 + 0] = a0; out[n * 3 + 1] = a1; out[n * 3 + 2] = a2;
}

__global__ void graph_out_kernel(const float* __restrict__ h, const float* __restrict__ lw,
                                 const float* __restrict__ lat, float* __restrict__ out)
{
  __shared__ float gf[HH];
  __shared__ float lm[9];
  int g = blockIdx.x, c = threadIdx.x;
  float s = 0.f;
  for (int i = 0; i < AA; ++i) s += h[(size_t)(g * AA + i) * HH + c];
  gf[c] = s / (float)AA;
  __syncthreads();
  if (c < 9) {
    float acc = 0.f;
    for (int k = 0; k < HH; ++k) acc += gf[k] * lw[k * 9 + c];
    lm[c] = acc;
  }
  __syncthreads();
  if (c < 9) {
    int i = c / 3, kk = c % 3;
    const float* L = lat + (size_t)g * 9;
    out[g * 9 + c] = lm[i*3+0]*L[0*3+kk] + lm[i*3+1]*L[1*3+kk] + lm[i*3+2]*L[2*3+kk];
  }
}

// ------------------------------- launcher ----------------------------------
extern "C" void kernel_launch(void* const* d_in, const int* in_sizes, int n_in,
                              void* d_out, int out_size, void* d_ws, size_t ws_size,
                              hipStream_t stream)
{
  (void)in_sizes; (void)n_in; (void)out_size; (void)ws_size;
  const float* t        = (const float*)d_in[0];
  const int*   at       = (const int*)  d_in[1];
  const float* frac     = (const float*)d_in[2];
  const float* lattices = (const float*)d_in[3];
  // d_in[4]=num_atoms, d_in[5]=node2graph: shapes are static (A=24 per graph)
  const float* xrd      = (const float*)d_in[6];
  const float* emb      = (const float*)d_in[7];
  const float* lw       = (const float*)d_in[8];
  const float* lb       = (const float*)d_in[9];
  const float* ew1      = (const float*)d_in[10];
  const float* eb1      = (const float*)d_in[11];
  const float* ew2      = (const float*)d_in[12];
  const float* eb2      = (const float*)d_in[13];
  const float* nw1      = (const float*)d_in[14];
  const float* nb1      = (const float*)d_in[15];
  const float* nw2      = (const float*)d_in[16];
  const float* nb2      = (const float*)d_in[17];
  const float* cw       = (const float*)d_in[18];
  const float* latw     = (const float*)d_in[19];

  char* ws = (char*)d_ws;
  size_t off = 0;
  float*    h      = (float*)   (ws + off); off += (size_t)NN * HH * 4;
  _Float16* fdemb  = (_Float16*)(ws + off); off += (size_t)EE * 64 * 2;
  float*    latips = (float*)   (ws + off); off += (size_t)GG * 9 * 4;
  _Float16* W1t    = (_Float16*)(ws + off); off += (size_t)4 * HH * K1PAD * 2;
  _Float16* W2t    = (_Float16*)(ws + off); off += (size_t)4 * HH * HH * 2;
  _Float16* N1t    = (_Float16*)(ws + off); off += (size_t)4 * HH * 256 * 2;
  _Float16* N2t    = (_Float16*)(ws + off); off += (size_t)4 * HH * HH * 2;

  latent_kernel<<<NN, HH, 0, stream>>>(t, at, xrd, emb, lw, lb, h);
  fdemb_kernel<<<(EE + 255) / 256, 256, 0, stream>>>(frac, fdemb);
  latips_kernel<<<(GG * 9 + 255) / 256, 256, 0, stream>>>(lattices, latips);
  conv_w1_kernel<<<(4 * HH * K1PAD + 255) / 256, 256, 0, stream>>>(ew1, W1t);
  conv_wt_kernel<<<(4 * HH * HH + 255) / 256, 256, 0, stream>>>(ew2, W2t, HH);
  conv_wt_kernel<<<(4 * HH * 256 + 255) / 256, 256, 0, stream>>>(nw1, N1t, 256);
  conv_wt_kernel<<<(4 * HH * HH + 255) / 256, 256, 0, stream>>>(nw2, N2t, HH);

  for (int l = 0; l < 4; ++l) {
    layer_kernel<<<GG, 256, 0, stream>>>(
        h, fdemb, latips,
        W1t + (size_t)l * HH * K1PAD, eb1 + l * HH,
        W2t + (size_t)l * HH * HH,    eb2 + l * HH,
        N1t + (size_t)l * HH * 256,   nb1 + l * HH,
        N2t + (size_t)l * HH * HH,    nb2 + l * HH);
  }

  float* out = (float*)d_out;
  graph_out_kernel<<<GG, HH, 0, stream>>>(h, latw, lattices, out);
  coord_kernel<<<(NN + 127) / 128, 128, 0, stream>>>(h, cw, out + GG * 9);
}
